// DynamicConvolution_75557064671830
// MI455X (gfx1250) — matrix operand order repaired and checked by reading, not compiled, not run on hardware
//
#include <hip/hip_runtime.h>
#include <math.h>

typedef _Float16 v16h __attribute__((ext_vector_type(16)));
typedef _Float16 v8h  __attribute__((ext_vector_type(8)));
typedef float    v8f  __attribute__((ext_vector_type(8)));
typedef unsigned int v4u __attribute__((ext_vector_type(4)));
typedef int v8i __attribute__((ext_vector_type(8)));
typedef int v4i __attribute__((ext_vector_type(4)));

#define BATCH 32
#define CIN   128
#define HDIM  64
#define WDIM  64
#define KNUM  4
#define OCH   256
#define RED   32
#define KSZ2  9
#define CK    (CIN * KSZ2)   // 1152 ; K ordered tap-major: k = tap*128 + c
#define HW    (HDIM * WDIM)  // 4096
#define LDT   40             // padded LDS row stride in f16 (80 B)
#define BM    128
#define BN    128

// ---------------- Kernel 1: pooled -> MLP -> softmax attention [B, K] ----------
__global__ __launch_bounds__(256) void attn_kernel(
    const float* __restrict__ x,
    const float* __restrict__ w1, const float* __restrict__ b1,
    const float* __restrict__ w2, const float* __restrict__ b2,
    float* __restrict__ att) {
  __shared__ float pooled[CIN];
  __shared__ float hbuf[RED];
  __shared__ float red[256];
  const int b = blockIdx.x;
  const int t = threadIdx.x;
  const float* xb = x + (size_t)b * CIN * HW;
  for (int c = 0; c < CIN; ++c) {
    const float* xc = xb + (size_t)c * HW;
    float s = 0.f;
    for (int i = t; i < HW; i += 256) s += xc[i];
    red[t] = s;
    __syncthreads();
    for (int off = 128; off > 0; off >>= 1) {
      if (t < off) red[t] += red[t + off];
      __syncthreads();
    }
    if (t == 0) pooled[c] = red[0] * (1.0f / HW);
    __syncthreads();
  }
  if (t < RED) {
    float s = b1[t];
    for (int c = 0; c < CIN; ++c) s += pooled[c] * w1[t * CIN + c];
    hbuf[t] = s > 0.f ? s : 0.f;
  }
  __syncthreads();
  if (t == 0) {
    float lg[KNUM];
    float mx = -1e30f;
    for (int k = 0; k < KNUM; ++k) {
      float s = b2[k];
      for (int r = 0; r < RED; ++r) s += hbuf[r] * w2[k * RED + r];
      lg[k] = s;
      mx = fmaxf(mx, s);
    }
    float den = 0.f;
    for (int k = 0; k < KNUM; ++k) { lg[k] = expf(lg[k] - mx); den += lg[k]; }
    for (int k = 0; k < KNUM; ++k) att[b * KNUM + k] = lg[k] / den;
  }
}

// ---- Kernel 2: mix kernels -> f16, tap-major K:  wk[b][o][tap*128 + c] --------
__global__ __launch_bounds__(256) void mix_kernel(
    const float* __restrict__ weights,   // [K][O][C][3][3]
    const float* __restrict__ att,       // [B][K]
    _Float16* __restrict__ wk) {         // [B][O][9][128]
  const size_t total = (size_t)BATCH * OCH * CK;
  size_t idx = (size_t)blockIdx.x * 256 + threadIdx.x;
  if (idx >= total) return;
  const int c = (int)(idx & 127);
  size_t r = idx >> 7;
  const int tap = (int)(r % KSZ2);
  r /= KSZ2;
  const int o = (int)(r % OCH);
  const int b = (int)(r / OCH);
  const size_t win   = ((size_t)o * CIN + c) * KSZ2 + tap;  // [o][c][tap]
  const size_t per_k = (size_t)OCH * CIN * KSZ2;
  const float a0 = att[b * KNUM + 0], a1 = att[b * KNUM + 1];
  const float a2 = att[b * KNUM + 2], a3 = att[b * KNUM + 3];
  float v = a0 * weights[win] + a1 * weights[per_k + win] +
            a2 * weights[2 * per_k + win] + a3 * weights[3 * per_k + win];
  wk[idx] = (_Float16)v;
}

// ------- Kernel 3: implicit-GEMM conv, WMMA + double-buffered TDM --------------
// out[b][O][HW] = wk[b][O][CK] x im2col(x[b])[CK][HW], K tap-major.
// Block: 256 threads (8 waves), tile M=128 x N=128, K-step 32, 2-deep pipeline.
__global__ __launch_bounds__(256) void conv_wmma_kernel(
    const float* __restrict__ x,
    const _Float16* __restrict__ wk,
    float* __restrict__ out) {
  __shared__ _Float16 As[2][BM * LDT];  // [m][k] padded, filled by TDM
  __shared__ _Float16 Bs[2][BN * LDT];  // [n][k] padded (col-major im2col tile)

  const int t    = threadIdx.x;
  const int lane = t & 31;
  const int wave = t >> 5;
  const int wm   = wave & 3;   // M quadrant: wm*32
  const int wn   = wave >> 2;  // N half:     wn*64
  const int grp  = lane >> 4;  // K-half selector per ISA layout
  const int lr   = lane & 15;

  const int b      = blockIdx.z;
  const int blockM = blockIdx.y * BM;
  const int blockN = blockIdx.x * BN;

  const float* xb = x + (size_t)b * CIN * HW;
  const size_t wk_row0 = ((size_t)b * OCH + blockM) * CK;  // element idx of tile row 0

  // ---- TDM descriptor (constant parts): 2D tile 32 x 128 f16 ----
  // group1: data_size=2B, pad_enable, pad_interval=16 dwords (64B), pad_amount=4 dwords (16B)
  v8i g1;
  g1[0] = (int)((1u << 16) | (1u << 20) | (3u << 22) | (3u << 25));
  g1[1] = (int)((32u & 0xffffu) << 16);   // tensor_dim0 = 32 (low 16 -> bits 63:48)
  g1[2] = (int)((128u & 0xffffu) << 16);  // tensor_dim1 = 128 (low 16 -> bits 95:80)
  g1[3] = (int)(32u << 16);               // tile_dim0 = 32 (bits 127:112)
  g1[4] = (int)128u;                      // tile_dim1 = 128, tile_dim2 = 0
  g1[5] = (int)(unsigned)CK;              // tensor_dim0_stride = 1152 (low 32)
  g1[6] = 0;
  g1[7] = 0;
  v4i gz4 = {0, 0, 0, 0};
  v8i gz8 = {0, 0, 0, 0, 0, 0, 0, 0};
  const unsigned lds_as0 = (unsigned)(uintptr_t)(&As[0][0]);
  const unsigned lds_as1 = (unsigned)(uintptr_t)(&As[1][0]);

  // TDM issue: one wave, DMA A tile for K-chunk kt into buffer `buf`
  auto tdm_issue = [&](int kt, int buf) {
    const unsigned long long ga =
        (unsigned long long)(uintptr_t)wk + (wk_row0 + (size_t)kt) * 2ull;
    v4u g0;
    g0[0] = 1u;                                  // count=1, user descriptor
    g0[1] = buf ? lds_as1 : lds_as0;             // LDS byte address
    g0[2] = (unsigned)(ga & 0xffffffffu);        // global_addr[31:0]
    g0[3] = (unsigned)(ga >> 32) | (2u << 30);   // global_addr[56:32] | type=2
    __builtin_amdgcn_tensor_load_to_lds(g0, g1, gz4, gz4, gz8, 0);
  };

  // B staging: thread handles spatial n = t&127, channels kk0..kk0+15
  const int bn  = t & 127;
  const int kk0 = (t >> 7) * 16;
  const int ohn = (blockN + bn) >> 6;
  const int own = bn & 63;
  auto stage_B = [&](int kt, int buf) {
    const int tap = kt >> 7;               // 128 % 32 == 0 -> single tap per chunk
    const int dh  = tap / 3 - 1;           // uniform scalars
    const int dw  = tap - (tap / 3) * 3 - 1;
    const int c0  = kt & 127;
    const int ih  = ohn + dh;
    const int iw  = own + dw;
    const bool ok = ((unsigned)ih < (unsigned)HDIM) & ((unsigned)iw < (unsigned)WDIM);
    const float* px = xb + (ih << 6) + iw;
    v8h vlo, vhi;
#pragma unroll
    for (int j = 0; j < 8; ++j) {
      float v0 = 0.f, v1 = 0.f;
      if (ok) {
        v0 = px[(size_t)(c0 + kk0 + j) << 12];
        v1 = px[(size_t)(c0 + kk0 + 8 + j) << 12];
      }
      vlo[j] = (_Float16)v0;
      vhi[j] = (_Float16)v1;
    }
    v8h* dst = reinterpret_cast<v8h*>(&Bs[buf][bn * LDT + kk0]);
    dst[0] = vlo;
    dst[1] = vhi;
  };

  v8f acc[2][4] = {};

  // ---- pipeline prologue: stage K-chunk 0 into buffer 0 ----
  if (wave == 0) tdm_issue(0, 0);
  stage_B(0, 0);
  if (wave == 0) __builtin_amdgcn_s_wait_tensorcnt(0);
  __syncthreads();

  int cur = 0;
  for (int kt = 0; kt < CK; kt += 32) {
    const int nxt = cur ^ 1;
    // ---- prefetch next K-chunk into the other buffer (overlaps WMMA) ----
    if (kt + 32 < CK) {
      if (wave == 0) tdm_issue(kt + 32, nxt);
      stage_B(kt + 32, nxt);
    }

    // ---- fragments (b128 pairs straight into fragment registers) ----
    union HV { v16h v; v8h h[2]; };
    HV af[2];  // A lane half g holds K in {g*8..+7} U {16+g*8..+7}
#pragma unroll
    for (int mi = 0; mi < 2; ++mi) {
      const v8h* pa =
          reinterpret_cast<const v8h*>(&As[cur][(wm * 32 + mi * 16 + lr) * LDT + grp * 8]);
      af[mi].h[0] = pa[0];
      af[mi].h[1] = pa[2];  // +16 f16
    }
    HV bf[4];  // B lane half g holds K in {g*8..+7} U {16+g*8..+7}
#pragma unroll
    for (int ni = 0; ni < 4; ++ni) {
      const v8h* pb =
          reinterpret_cast<const v8h*>(&Bs[cur][(wn * 64 + ni * 16 + lr) * LDT + grp * 8]);
      bf[ni].h[0] = pb[0];
      bf[ni].h[1] = pb[2];
    }

#pragma unroll
    for (int mi = 0; mi < 2; ++mi)
#pragma unroll
      for (int ni = 0; ni < 4; ++ni)
        acc[mi][ni] = __builtin_amdgcn_wmma_f32_16x16x32_f16(
            false, af[mi].v, false, bf[ni].v, (short)0, acc[mi][ni], false, false);

    // ---- close pipeline stage: TDM drained, gathers + compute fenced ----
    if (wave == 0) __builtin_amdgcn_s_wait_tensorcnt(0);
    __syncthreads();
    cur = nxt;
  }

  // ---- epilogue: C/D layout -> lane half grp holds rows M = grp*8 + v ----
  // Output is write-once: non-temporal stores keep L2 for x/wk reuse.
#pragma unroll
  for (int mi = 0; mi < 2; ++mi) {
    const int orow = blockM + wm * 32 + mi * 16 + grp * 8;
#pragma unroll
    for (int ni = 0; ni < 4; ++ni) {
      const int col = blockN + wn * 64 + ni * 16 + lr;
      float* dst = out + ((size_t)b * OCH + orow) * HW + col;
#pragma unroll
      for (int v = 0; v < 8; ++v)
        __builtin_nontemporal_store(acc[mi][ni][v], dst + (size_t)v * HW);
    }
  }
}

// ---------------- launcher -----------------------------------------------------
extern "C" void kernel_launch(void* const* d_in, const int* in_sizes, int n_in,
                              void* d_out, int out_size, void* d_ws, size_t ws_size,
                              hipStream_t stream) {
  const float* x       = (const float*)d_in[0];
  const float* weights = (const float*)d_in[1];
  const float* w1      = (const float*)d_in[2];
  const float* b1      = (const float*)d_in[3];
  const float* w2      = (const float*)d_in[4];
  const float* b2      = (const float*)d_in[5];
  float* out = (float*)d_out;

  float*    att = (float*)d_ws;                       // 32*4 floats
  _Float16* wkf = (_Float16*)((char*)d_ws + 1024);    // [B][O][9][128] f16, ~18.9 MB

  attn_kernel<<<BATCH, 256, 0, stream>>>(x, w1, b1, w2, b2, att);

  const size_t nmix = (size_t)BATCH * OCH * CK;
  mix_kernel<<<(unsigned)((nmix + 255) / 256), 256, 0, stream>>>(weights, att, wkf);

  dim3 grid(HW / BN, OCH / BM, BATCH);
  conv_wmma_kernel<<<grid, 256, 0, stream>>>(x, wkf, out);
}
